// BetaMoE_87625922773494
// MI455X (gfx1250) — matrix operand order repaired and checked by reading, not compiled, not run on hardware
//
#include <hip/hip_runtime.h>
#include <hip/hip_bf16.h>

// ---------------- problem dims ----------------
constexpr int BATCH = 256;
constexpr int TSEQ  = 60;
constexpr int TPAD  = 64;     // padded time for 16x16 tiling
constexpr int CDIM  = 512;
constexpr int NEXP  = 10;
constexpr int KSEL  = 5;
constexpr int NH    = 8;
constexpr int DHEAD = 64;
constexpr int CHALF = 256;    // C/2
constexpr int KVCOLS = 1024;  // K||V fused projection width

typedef _Float16 h16;
typedef _Float16 v8h  __attribute__((ext_vector_type(8)));
typedef _Float16 v16h __attribute__((ext_vector_type(16)));
typedef float    v8f  __attribute__((ext_vector_type(8)));
typedef int      v4i  __attribute__((ext_vector_type(4)));

// ---- CDNA5 async global->LDS path (guarded; falls back to plain LDS staging) ----
#if defined(__has_builtin)
#  if __has_builtin(__builtin_amdgcn_global_load_async_to_lds_b128)
#    define HAVE_ASYNC_LDS 1
#  endif
#endif
#ifndef HAVE_ASYNC_LDS
#  define HAVE_ASYNC_LDS 0
#endif

__device__ __forceinline__ void async_stage_b128(const h16* g, h16* l) {
#if HAVE_ASYNC_LDS
  typedef __attribute__((address_space(1))) void* as1vp;
  typedef __attribute__((address_space(3))) void* as3vp;
  typedef __attribute__((address_space(1))) v4i* as1p;
  typedef __attribute__((address_space(3))) v4i* as3p;
  as1p gp = (as1p)(as1vp)(void*)const_cast<h16*>(g);
  as3p lp = (as3p)(as3vp)(void*)l;
  __builtin_amdgcn_global_load_async_to_lds_b128(gp, lp, 0, 0);
#else
  *(v8h*)l = *(const v8h*)g;
#endif
}

__device__ __forceinline__ void wait_async_le1() {
#if HAVE_ASYNC_LDS
#  if __has_builtin(__builtin_amdgcn_s_wait_asynccnt)
  __builtin_amdgcn_s_wait_asynccnt(1);
#  else
  asm volatile("s_wait_asynccnt 0x1" ::: "memory");
#  endif
#endif
}

__device__ __forceinline__ void wait_async_le0() {
#if HAVE_ASYNC_LDS
#  if __has_builtin(__builtin_amdgcn_s_wait_asynccnt)
  __builtin_amdgcn_s_wait_asynccnt(0);
#  else
  asm volatile("s_wait_asynccnt 0x0" ::: "memory");
#  endif
#endif
}

__device__ __forceinline__ float softplusf(float x) {
  return (x > 0.f) ? x + log1pf(__expf(-x)) : log1pf(__expf(x));
}

// Build 16x32 f16 A-fragment for lane (M = lane%16).
// Layout: half==0 -> K {kb..kb+15}; half==1 -> {kb+16..kb+31} (matches contiguous B-frag)
__device__ __forceinline__ v16h load_afrag(const h16* __restrict__ A, int row, int kb, int half) {
  v8h a0 = *(const v8h*)(A + (size_t)row * CDIM + kb + 16 * half);
  v8h a1 = *(const v8h*)(A + (size_t)row * CDIM + kb + 16 * half + 8);
  v16h a;
#pragma unroll
  for (int i = 0; i < 8; ++i) { a[i] = a0[i]; a[8 + i] = a1[i]; }
  return a;
}

// ---------------- prep: fp32 -> f16 conversions ----------------
__global__ void prep_data_kernel(const float* __restrict__ data, h16* __restrict__ out) {
  int total = BATCH * TPAD * CDIM;
  for (int idx = blockIdx.x * blockDim.x + threadIdx.x; idx < total; idx += gridDim.x * blockDim.x) {
    int c = idx & (CDIM - 1);
    int r = (idx / CDIM) & (TPAD - 1);
    int b = idx / (CDIM * TPAD);
    float v = (r < TSEQ) ? data[((size_t)b * TSEQ + r) * CDIM + c] : 0.f;
    out[idx] = (h16)v;
  }
}

__global__ void cvt_kernel(const float* __restrict__ src, h16* __restrict__ dst, int n) {
  int idx = blockIdx.x * blockDim.x + threadIdx.x;
  if (idx < n) dst[idx] = (h16)src[idx];
}

// ---------------- K/V projection GEMM (WMMA f16) ----------------
// Output kv16[b*64 + t][0..511]=K-proj, [512..1023]=V-proj.
__global__ __launch_bounds__(256) void kv_gemm_kernel(
    const h16* __restrict__ data16, const h16* __restrict__ wkv16,
    const float* __restrict__ in_proj_b, h16* __restrict__ kv16) {
  int nb = blockIdx.x;   // 0..7  (128-wide N blocks)
  int mb = blockIdx.y;   // 0..255 (one batch = one 64-row M block)
  int wave = threadIdx.x >> 5;
  int lane = threadIdx.x & 31;
  int half = lane >> 4;
  int nidx = lane & 15;
  int mt = wave & 3;           // M tile (4 per block)
  int ntb = (wave >> 2) * 4;   // 4 N tiles per wave

  const h16* A = data16 + (size_t)mb * TPAD * CDIM;
  int arow = 16 * mt + nidx;

  v8f acc[4];
#pragma unroll
  for (int j = 0; j < 4; ++j)
#pragma unroll
    for (int i = 0; i < 8; ++i) acc[j][i] = 0.f;

  for (int ks = 0; ks < 16; ++ks) {
    int kb = ks * 32;
    v16h af = load_afrag(A, arow, kb, half);
#pragma unroll
    for (int j = 0; j < 4; ++j) {
      int ncol = nb * 128 + (ntb + j) * 16 + nidx;
      v16h bf = *(const v16h*)(wkv16 + (size_t)ncol * CDIM + kb + 16 * half);
      acc[j] = __builtin_amdgcn_wmma_f32_16x16x32_f16(
          false, af, false, bf, (short)0, acc[j], false, false);
    }
  }
#pragma unroll
  for (int j = 0; j < 4; ++j) {
    int ncol = nb * 128 + (ntb + j) * 16 + nidx;
    float bias = in_proj_b[CDIM + ncol];  // rows C..3C of in_proj
#pragma unroll
    for (int r = 0; r < 8; ++r) {
      int rowg = mb * TPAD + 16 * mt + 8 * half + r;
      kv16[(size_t)rowg * KVCOLS + ncol] = (h16)(acc[j][r] + bias);
    }
  }
}

// ---------------- attention + router + beta weights (one block per batch) ----------------
__global__ __launch_bounds__(256) void att_kernel(
    const float* __restrict__ qst, const float* __restrict__ in_proj_w,
    const float* __restrict__ in_proj_b, const h16* __restrict__ kv16,
    const float* __restrict__ out_proj_w, const float* __restrict__ out_proj_b,
    const float* __restrict__ router_w, const float* __restrict__ router_b,
    const float* __restrict__ beta_w, const float* __restrict__ beta_b,
    int* __restrict__ inds_g, float* __restrict__ probs_g,
    float* __restrict__ tw_g, float* __restrict__ sumtw_g) {
  int b = blockIdx.x, tid = threadIdx.x;
  __shared__ float qv[CDIM];
  __shared__ float sc[NH][TPAD];
  __shared__ float ctxs[CDIM];
  __shared__ float tmpw[CDIM];
  __shared__ float rl[NEXP];
  __shared__ float bp[2 * NEXP];
  __shared__ int   sel[KSEL];
  __shared__ float wb[KSEL][TPAD];

  const float* qrow = qst + (size_t)b * CDIM;
  for (int c = tid; c < CDIM; c += 256) {
    const float* w = in_proj_w + (size_t)c * CDIM;  // wq rows 0..C-1
    float s = 0.f;
    for (int j = 0; j < CDIM; ++j) s += qrow[j] * w[j];
    qv[c] = (s + in_proj_b[c]) * 0.125f;  // 1/sqrt(DH)
  }
  __syncthreads();
  for (int idx = tid; idx < NH * TSEQ; idx += 256) {
    int h = idx / TSEQ, t = idx % TSEQ;
    const h16* krow = kv16 + ((size_t)b * TPAD + t) * KVCOLS + h * DHEAD;
    float s = 0.f;
    for (int d = 0; d < DHEAD; ++d) s += qv[h * DHEAD + d] * (float)krow[d];
    sc[h][t] = s;
  }
  __syncthreads();
  if (tid < NH) {
    float mx = -1e30f;
    for (int t = 0; t < TSEQ; ++t) mx = fmaxf(mx, sc[tid][t]);
    float s = 0.f;
    for (int t = 0; t < TSEQ; ++t) { float e = __expf(sc[tid][t] - mx); sc[tid][t] = e; s += e; }
    float inv = 1.f / s;
    for (int t = 0; t < TSEQ; ++t) sc[tid][t] *= inv;
  }
  __syncthreads();
  for (int c = tid; c < CDIM; c += 256) {
    int h = c / DHEAD;
    float s = 0.f;
    for (int t = 0; t < TSEQ; ++t)
      s += sc[h][t] * (float)kv16[((size_t)b * TPAD + t) * KVCOLS + CDIM + c];
    ctxs[c] = s;
  }
  __syncthreads();
  for (int c = tid; c < CDIM; c += 256) {
    const float* w = out_proj_w + (size_t)c * CDIM;
    float s = 0.f;
    for (int j = 0; j < CDIM; ++j) s += ctxs[j] * w[j];
    tmpw[c] = s + out_proj_b[c];
  }
  __syncthreads();
  if (tid < NEXP) {
    const float* w = router_w + (size_t)tid * CDIM;
    float s = 0.f;
    for (int j = 0; j < CDIM; ++j) s += tmpw[j] * w[j];
    rl[tid] = s + router_b[tid];
  }
  if (tid >= 32 && tid < 32 + 2 * NEXP) {
    int i = tid - 32;
    const float* w = beta_w + (size_t)i * CDIM;
    float s = 0.f;
    for (int j = 0; j < CDIM; ++j) s += tmpw[j] * w[j];
    bp[i] = s + beta_b[i];
  }
  __syncthreads();
  if (tid == 0) {
    float mx = -1e30f;
    for (int e = 0; e < NEXP; ++e) mx = fmaxf(mx, rl[e]);
    float rp[NEXP]; float s = 0.f;
    for (int e = 0; e < NEXP; ++e) { rp[e] = __expf(rl[e] - mx); s += rp[e]; }
    float inv = 1.f / s;
    for (int e = 0; e < NEXP; ++e) rp[e] *= inv;
    bool used[NEXP] = {};
    float sp[KSEL]; float psum = 0.f;
    for (int kk = 0; kk < KSEL; ++kk) {
      int best = 0; float bv = -1.f;
      for (int e = 0; e < NEXP; ++e)
        if (!used[e] && rp[e] > bv) { bv = rp[e]; best = e; }
      used[best] = true; sel[kk] = best; sp[kk] = bv; psum += bv;
    }
    float pin = 1.f / (psum + 1e-8f);
    for (int kk = 0; kk < KSEL; ++kk) {
      inds_g[b * KSEL + kk] = sel[kk];
      probs_g[b * KSEL + kk] = sp[kk] * pin;
    }
  }
  __syncthreads();
  for (int idx = tid; idx < KSEL * TSEQ; idx += 256) {
    int kk = idx / TSEQ, t = idx % TSEQ;
    int e = sel[kk];
    float a  = softplusf(bp[2 * e])     + 1e-6f;
    float bb = softplusf(bp[2 * e + 1]) + 1e-6f;
    float tv = (float)t * (1.f / (float)(TSEQ - 1));
    float lt = __logf(tv + 1e-12f);
    float l1 = __logf(1.f - tv + 1e-12f);
    float lp = (a - 1.f) * lt + (bb - 1.f) * l1
             - (lgammaf(a) + lgammaf(bb) - lgammaf(a + bb));
    wb[kk][t] = __expf(lp);
  }
  __syncthreads();
  if (tid < KSEL) {
    float mx = 0.f;
    for (int t = 0; t < TSEQ; ++t) mx = fmaxf(mx, wb[tid][t]);
    float inv = 1.f / (mx + 1e-8f);
    float s = 0.f;
    float* dst = tw_g + ((size_t)b * KSEL + tid) * TPAD;
    for (int t = 0; t < TSEQ; ++t) { float v = wb[tid][t] * inv; dst[t] = v; s += v; }
    for (int t = TSEQ; t < TPAD; ++t) dst[t] = 0.f;  // kills padded GEMM rows
    sumtw_g[b * KSEL + tid] = s;
  }
}

// ---------------- MoE: h=relu(data@w1^T+b1), pool over t, then pooled@w2^T ----------------
// A-tile (64 x 32 k-chunk) is staged into LDS via async global->LDS DMA, double buffered,
// so the WMMA stream overlaps the activation fetch; weights stream from global as B-frags.
__global__ __launch_bounds__(256) void moe_kernel(
    const h16* __restrict__ data16, const h16* __restrict__ w1_16,
    const h16* __restrict__ w2_16, const float* __restrict__ exp_b1,
    const float* __restrict__ exp_b2, const int* __restrict__ inds_g,
    const float* __restrict__ probs_g, const float* __restrict__ tw_g,
    const float* __restrict__ sumtw_g, float* __restrict__ accum) {
  int bk = blockIdx.x;
  int b = bk / KSEL, kk = bk % KSEL;
  int e = inds_g[b * KSEL + kk];
  float p = probs_g[b * KSEL + kk];

  __shared__ float twl[TPAD];
  __shared__ float pooled[CHALF];
  __shared__ __align__(16) h16 ldsA[2][TPAD * 32];  // double-buffered 64x32 A chunk
  int tid = threadIdx.x;
  if (tid < TPAD) twl[tid] = tw_g[((size_t)b * KSEL + kk) * TPAD + tid];
  pooled[tid] = 0.f;  // blockDim 256 == CHALF
  __syncthreads();

  int wave = tid >> 5, lane = tid & 31, half = lane >> 4, nidx = lane & 15;
  int mt = wave & 3;            // 4 M tiles (t = 0..63)
  int ntb = (wave >> 2) * 8;    // 8 N tiles per wave (N = 256)
  const h16* A  = data16 + (size_t)b * TPAD * CDIM;
  const h16* W1 = w1_16 + (size_t)e * CHALF * CDIM;
  int arow = 16 * mt + nidx;

  // staging map: thread -> (row, 8-elem segment) of the 64x32 chunk, 16B per thread
  const int srow = tid >> 2;
  const int sseg = (tid & 3) * 8;

  v8f acc[8];
#pragma unroll
  for (int j = 0; j < 8; ++j)
#pragma unroll
    for (int i = 0; i < 8; ++i) acc[j][i] = 0.f;

  // prologue: stage chunk 0
  async_stage_b128(A + (size_t)srow * CDIM + sseg, &ldsA[0][srow * 32 + sseg]);

  for (int ks = 0; ks < 16; ++ks) {
    int buf = ks & 1;
    if (ks < 15) {
      async_stage_b128(A + (size_t)srow * CDIM + (ks + 1) * 32 + sseg,
                       &ldsA[buf ^ 1][srow * 32 + sseg]);
      wait_async_le1();   // chunk ks complete (this wave's op; loads retire in order)
    } else {
      wait_async_le0();
    }
    __syncthreads();      // all waves' chunk-ks stages visible

    const h16* Ab = &ldsA[buf][0];
    v8h a0 = *(const v8h*)(Ab + arow * 32 + 16 * half);
    v8h a1 = *(const v8h*)(Ab + arow * 32 + 16 * half + 8);
    v16h af;
#pragma unroll
    for (int i = 0; i < 8; ++i) { af[i] = a0[i]; af[8 + i] = a1[i]; }

    int kb = ks * 32;
#pragma unroll
    for (int j = 0; j < 8; ++j) {
      int ncol = (ntb + j) * 16 + nidx;
      v16h bf = *(const v16h*)(W1 + (size_t)ncol * CDIM + kb + 16 * half);
      acc[j] = __builtin_amdgcn_wmma_f32_16x16x32_f16(
          false, af, false, bf, (short)0, acc[j], false, false);
    }
    __syncthreads();      // protect buf before it is overwritten next iter
  }

  // fused bias + ReLU + time-pooling on D-fragment layout (M = 16*mt + 8*half + r)
#pragma unroll
  for (int j = 0; j < 8; ++j) {
    int ncol = (ntb + j) * 16 + nidx;
    float b1v = exp_b1[e * CHALF + ncol];
    float part = 0.f;
#pragma unroll
    for (int r = 0; r < 8; ++r) {
      int t = 16 * mt + 8 * half + r;
      float hv = fmaxf(acc[j][r] + b1v, 0.f);
      part += twl[t] * hv;
    }
    atomicAdd(&pooled[ncol], part);  // ds_add_f32 cross-wave reduction
  }
  __syncthreads();

  float stw = sumtw_g[b * KSEL + kk];
  for (int c = tid; c < CDIM; c += 256) {
    const h16* w2r = w2_16 + ((size_t)e * CDIM + c) * CHALF;
    float s = 0.f;
    for (int j = 0; j < CHALF; ++j) s += pooled[j] * (float)w2r[j];
    float outv = p * (s + stw * exp_b2[e * CDIM + c]);
    atomicAdd(&accum[(size_t)b * CDIM + c], outv);
  }
}

// ---------------- LayerNorm ----------------
__global__ __launch_bounds__(256) void ln_kernel(
    const float* __restrict__ accum, const float* __restrict__ g,
    const float* __restrict__ bt, float* __restrict__ out) {
  int b = blockIdx.x, tid = threadIdx.x;
  __shared__ float rs[256], rq[256];
  float x0 = accum[(size_t)b * CDIM + tid];
  float x1 = accum[(size_t)b * CDIM + tid + 256];
  rs[tid] = x0 + x1;
  rq[tid] = x0 * x0 + x1 * x1;
  __syncthreads();
  for (int off = 128; off > 0; off >>= 1) {
    if (tid < off) { rs[tid] += rs[tid + off]; rq[tid] += rq[tid + off]; }
    __syncthreads();
  }
  float mu  = rs[0] * (1.f / CDIM);
  float var = rq[0] * (1.f / CDIM) - mu * mu;
  float inv = rsqrtf(var + 1e-5f);
  out[(size_t)b * CDIM + tid]       = (x0 - mu) * inv * g[tid] + bt[tid];
  out[(size_t)b * CDIM + tid + 256] = (x1 - mu) * inv * g[tid + 256] + bt[tid + 256];
}

// ---------------- launch ----------------
extern "C" void kernel_launch(void* const* d_in, const int* in_sizes, int n_in,
                              void* d_out, int out_size, void* d_ws, size_t ws_size,
                              hipStream_t stream) {
  (void)in_sizes; (void)n_in; (void)out_size; (void)ws_size;
  const float* qst        = (const float*)d_in[0];
  const float* data       = (const float*)d_in[1];
  const float* in_proj_w  = (const float*)d_in[2];
  const float* in_proj_b  = (const float*)d_in[3];
  const float* out_proj_w = (const float*)d_in[4];
  const float* out_proj_b = (const float*)d_in[5];
  const float* router_w   = (const float*)d_in[6];
  const float* router_b   = (const float*)d_in[7];
  const float* beta_w     = (const float*)d_in[8];
  const float* beta_b     = (const float*)d_in[9];
  const float* exp_w1     = (const float*)d_in[10];
  const float* exp_b1     = (const float*)d_in[11];
  const float* exp_w2     = (const float*)d_in[12];
  const float* exp_b2     = (const float*)d_in[13];
  const float* ln_g       = (const float*)d_in[14];
  const float* ln_b       = (const float*)d_in[15];

  char* ws = (char*)d_ws;
  size_t off = 0;
  auto alloc = [&](size_t bytes) -> void* {
    void* p = ws + off;
    off += (bytes + 255) & ~(size_t)255;
    return p;
  };
  h16*   data16  = (h16*)alloc((size_t)BATCH * TPAD * CDIM * 2);
  h16*   kv16    = (h16*)alloc((size_t)BATCH * TPAD * KVCOLS * 2);
  h16*   wkv16   = (h16*)alloc((size_t)KVCOLS * CDIM * 2);
  h16*   w1_16   = (h16*)alloc((size_t)NEXP * CHALF * CDIM * 2);
  h16*   w2_16   = (h16*)alloc((size_t)NEXP * CDIM * CHALF * 2);
  int*   inds_g  = (int*)alloc((size_t)BATCH * KSEL * 4);
  float* probs_g = (float*)alloc((size_t)BATCH * KSEL * 4);
  float* tw_g    = (float*)alloc((size_t)BATCH * KSEL * TPAD * 4);
  float* sumtw_g = (float*)alloc((size_t)BATCH * KSEL * 4);
  float* accum   = (float*)alloc((size_t)BATCH * CDIM * 4);

  prep_data_kernel<<<4096, 256, 0, stream>>>(data, data16);
  cvt_kernel<<<(KVCOLS * CDIM + 255) / 256, 256, 0, stream>>>(in_proj_w + (size_t)CDIM * CDIM, wkv16, KVCOLS * CDIM);
  cvt_kernel<<<(NEXP * CHALF * CDIM + 255) / 256, 256, 0, stream>>>(exp_w1, w1_16, NEXP * CHALF * CDIM);
  cvt_kernel<<<(NEXP * CDIM * CHALF + 255) / 256, 256, 0, stream>>>(exp_w2, w2_16, NEXP * CDIM * CHALF);

  kv_gemm_kernel<<<dim3(8, BATCH), 256, 0, stream>>>(data16, wkv16, in_proj_b, kv16);

  att_kernel<<<BATCH, 256, 0, stream>>>(qst, in_proj_w, in_proj_b, kv16,
                                        out_proj_w, out_proj_b, router_w, router_b,
                                        beta_w, beta_b, inds_g, probs_g, tw_g, sumtw_g);

  (void)hipMemsetAsync(accum, 0, (size_t)BATCH * CDIM * 4, stream);

  moe_kernel<<<BATCH * KSEL, 256, 0, stream>>>(data16, w1_16, w2_16, exp_b1, exp_b2,
                                               inds_g, probs_g, tw_g, sumtw_g, accum);

  ln_kernel<<<BATCH, 256, 0, stream>>>(accum, ln_g, ln_b, (float*)d_out);
}
